// RNNDecoderWithAttention_12816182411641
// MI455X (gfx1250) — matrix operand order repaired and checked by reading, not compiled, not run on hardware
//
#include <hip/hip_runtime.h>
#include <hip/hip_bf16.h>

// ---------------------------------------------------------------------------
// RNN decoder with soft attention (show-attend-tell) for MI455X / gfx1250.
// - All GEMMs: v_wmma_f32_16x16x32_bf16 (bf16 A/B, fp32 accumulate).
// - Weights, enc, and per-step activations packed into bf16 WMMA-fragment-
//   major layout (32B/lane contiguous): inner loops are pure
//   global_load_b128 -> v_wmma, no converts.
// - Generic GEMM K-loop is software-pipelined (register double buffer):
//   kt+1 fragments are in flight while kt's WMMAs execute, so waits are
//   partial instead of s_wait_loadcnt 0 before every WMMA (M=64 -> few
//   waves, the wave itself must hide L2 latency).
// - enc_att GEMM stages the block-shared A tile through LDS with gfx1250
//   global_load_async_to_lds_b128 (double buffered, ASYNCcnt).
// - global_prefetch_b8 on the streamed weight tiles.
// ---------------------------------------------------------------------------

typedef __attribute__((ext_vector_type(16))) __bf16 v16bf;
typedef __attribute__((ext_vector_type(8)))  float  v8f;

#define BSZ   64
#define PPOS  196      // 14*14 spatial positions
#define DENC  2048
#define ADIM  512
#define EDIM  512
#define HDIM  512
#define VOC   10000
#define LCAP  24
#define TSTEP 23       // L - 1
#define XDIM  (EDIM + DENC)   // 2560
#define GDIM  (4 * HDIM)      // 2048

// Packed fragment: tile = (row_tile * ktiles + kt); lane-contiguous 32 bytes.
__device__ __forceinline__ v16bf load_packed(const __bf16* __restrict__ P,
                                             size_t tile, int lane) {
  return *(const v16bf*)(P + (tile * 32 + lane) * 16);
}

__device__ __forceinline__ float sigmoidf_(float x) {
  return 1.0f / (1.0f + __expf(-x));
}

// --------------------------- one-time repacking ----------------------------
// W (N x K row-major fp32) -> packed bf16 B-fragments.
// B frag element e of lane (h*16+n): e<8: k=h*8+e ; e>=8: k=16+h*8+(e-8).
__global__ void pack_w_kernel(const float* __restrict__ W,
                              __bf16* __restrict__ P, int N, int K) {
  const int gid  = blockIdx.x * blockDim.x + threadIdx.x;
  const int lane = gid & 31;
  const int tile = gid >> 5;
  const int ktiles = K >> 5;
  const int nt = tile / ktiles, kt = tile % ktiles;
  if (nt * 16 >= N) return;
  const int n = nt * 16 + (lane & 15);
  const int h = lane >> 4;
  const float* r = W + (size_t)n * K + kt * 32 + h * 8;
  __bf16* o = P + ((size_t)tile * 32 + lane) * 16;
#pragma unroll
  for (int e = 0; e < 8; ++e) { o[e] = (__bf16)r[e]; o[8 + e] = (__bf16)r[16 + e]; }
}

// A (M x K row-major fp32) -> packed bf16 A-fragments.
// A frag element e of lane (h*16+m): e<8: k=h*8+e ; e>=8: k=16+h*8+(e-8).
__global__ void pack_a_kernel(const float* __restrict__ A,
                              __bf16* __restrict__ P, int M, int K) {
  const int gid  = blockIdx.x * blockDim.x + threadIdx.x;
  const int lane = gid & 31;
  const int tile = gid >> 5;
  const int ktiles = K >> 5;
  const int mt = tile / ktiles, kt = tile % ktiles;
  if (mt * 16 >= M) return;
  const int m = mt * 16 + (lane & 15);
  const int h = lane >> 4;
  const float* r = A + (size_t)m * K + kt * 32 + h * 8;
  __bf16* o = P + ((size_t)tile * 32 + lane) * 16;
#pragma unroll
  for (int e = 0; e < 8; ++e) { o[e] = (__bf16)r[e]; o[8 + e] = (__bf16)r[16 + e]; }
}

// ------------------------- generic packed WMMA GEMM ------------------------
// C[M,N] = act( A[M,K] @ W[N,K]^T + bias ) (+C if accumulate); A,W packed
// bf16. Each wave owns 2 16x16 tiles sharing one A fragment; K-loop is
// software-pipelined one iteration ahead. act:0=id, 1=sigmoid.
// lens!=null -> row m zeroed unless t < lens[m]-1 (pred masking).
__global__ __launch_bounds__(128)
void gemm_wmma_pk(const __bf16* __restrict__ Ap,
                  const __bf16* __restrict__ Wp,
                  const float* __restrict__ bias,
                  float* __restrict__ C, int ldc,
                  int M, int N, int K, int act, int accumulate,
                  const long long* __restrict__ lens, int t) {
  const int lane  = threadIdx.x;
  const int wtile = blockIdx.x * blockDim.y + threadIdx.y;
  const int mt = blockIdx.y;
  const int m0 = mt * 16;
  const int nt0 = wtile * 2;
  if (m0 >= M || nt0 * 16 >= N) return;     // wave-uniform, EXEC stays full

  const int ktiles = K >> 5;
  const int n1ok = ((nt0 + 1) * 16 < N);
  const size_t at = (size_t)mt * ktiles;
  const size_t w0 = (size_t)nt0 * ktiles;
  const size_t w1 = (size_t)(n1ok ? nt0 + 1 : nt0) * ktiles;

  // prologue: fragments for kt = 0
  v16bf a  = load_packed(Ap, at, lane);
  v16bf b0 = load_packed(Wp, w0, lane);
  v16bf b1 = load_packed(Wp, w1, lane);

  v8f acc0 = {}, acc1 = {};
  for (int kt = 0; kt < ktiles; ++kt) {
    const int kn = (kt + 1 < ktiles) ? kt + 1 : kt;   // clamp: in-bounds reload
    // issue next iteration's fragments before consuming this one's
    v16bf an  = load_packed(Ap, at + kn, lane);
    v16bf bn0 = load_packed(Wp, w0 + kn, lane);
    v16bf bn1 = load_packed(Wp, w1 + kn, lane);
    if (kt + 4 < ktiles)   // gfx1250 global_prefetch_b8 on packed weight walk
      __builtin_prefetch(Wp + (w0 + kt + 4) * 512, 0, 1);
    acc0 = __builtin_amdgcn_wmma_f32_16x16x32_bf16(false, a, false, b0,
                                                   (short)0, acc0, false, false);
    acc1 = __builtin_amdgcn_wmma_f32_16x16x32_bf16(false, a, false, b1,
                                                   (short)0, acc1, false, false);
    a = an; b0 = bn0; b1 = bn1;
  }

  const int h = lane >> 4;
  const int nA = nt0 * 16 + (lane & 15);
  const int nB = (nt0 + 1) * 16 + (lane & 15);
#pragma unroll
  for (int r = 0; r < 8; ++r) {
    const int m = m0 + r + h * 8;           // C/D layout (ISA 7.12.2)
    const int msk = lens ? (t < (int)(lens[m] - 1)) : 1;
    float v = acc0[r] + (bias ? bias[nA] : 0.0f);
    if (accumulate) v += C[(size_t)m * ldc + nA];
    if (act == 1) v = sigmoidf_(v);
    if (!msk) v = 0.0f;
    C[(size_t)m * ldc + nA] = v;
    if (n1ok) {                             // wave-uniform
      float u = acc1[r] + (bias ? bias[nB] : 0.0f);
      if (accumulate) u += C[(size_t)m * ldc + nB];
      if (act == 1) u = sigmoidf_(u);
      if (!msk) u = 0.0f;
      C[(size_t)m * ldc + nB] = u;
    }
  }
}

// ---------------- enc_att GEMM with async-LDS A staging --------------------
// All 4 waves of a block share one A tile (same mt). Wave 0 stages the 1KB
// packed A tile into LDS via gfx1250 async copies (tracked by ASYNCcnt),
// double buffered against the WMMA consumption of the other buffer.
__global__ __launch_bounds__(128)
void gemm_wmma_pkA_async(const __bf16* __restrict__ Ap,
                         const __bf16* __restrict__ Wp,
                         const float* __restrict__ bias,
                         float* __restrict__ C, int ldc,
                         int M, int N, int K) {
  constexpr int NT = 2;
  __shared__ __align__(16) __bf16 sA[2][512];   // 2 x 1KB tile double buffer
  const int lane = threadIdx.x;
  const int wv   = threadIdx.y;                 // 0..3
  const int wtile = blockIdx.x * 4 + wv;
  const int mt = blockIdx.y;
  const int ktiles = K >> 5;
  const int nt0 = wtile * NT;
  const int maxnt = (N >> 4) - 1;
  int ntv[NT];
#pragma unroll
  for (int j = 0; j < NT; ++j) {
    int nt = nt0 + j;
    ntv[j] = nt <= maxnt ? nt : maxnt;          // clamp; no early-exit (barriers)
  }

  const __bf16* Abase = Ap + (size_t)mt * ktiles * 512;
  const unsigned lds0 = (unsigned)(size_t)(&sA[0][0]) + lane * 16;
  const unsigned lds1 = (unsigned)(size_t)(&sA[1][0]) + lane * 16;

  // prologue: stage kt = 0 into buffer 0
  if (wv == 0) {
    unsigned long long g = (unsigned long long)(size_t)Abase + lane * 16;
    asm volatile("global_load_async_to_lds_b128 %0, %1, off"
                 :: "v"(lds0), "v"(g) : "memory");
    asm volatile("global_load_async_to_lds_b128 %0, %1, off offset:512"
                 :: "v"(lds0), "v"(g) : "memory");
  }

  v8f acc[NT] = {};
  for (int kt = 0; kt < ktiles; ++kt) {
    const int cur = kt & 1;
    if (wv == 0)
      asm volatile("s_wait_asynccnt 0x0" ::: "memory");   // buf[cur] resident
    __syncthreads();
    if (wv == 0 && kt + 1 < ktiles) {                     // stage next tile
      unsigned long long g =
          (unsigned long long)(size_t)(Abase + (size_t)(kt + 1) * 512) + lane * 16;
      unsigned l = cur ? lds0 : lds1;
      asm volatile("global_load_async_to_lds_b128 %0, %1, off"
                   :: "v"(l), "v"(g) : "memory");
      asm volatile("global_load_async_to_lds_b128 %0, %1, off offset:512"
                   :: "v"(l), "v"(g) : "memory");
    }
    v16bf a = *(const v16bf*)(&sA[cur][lane * 16]);       // ds_load_b128 x2
#pragma unroll
    for (int j = 0; j < NT; ++j) {
      v16bf b = load_packed(Wp, (size_t)ntv[j] * ktiles + kt, lane);
      acc[j] = __builtin_amdgcn_wmma_f32_16x16x32_bf16(false, a, false, b,
                                                       (short)0, acc[j], false, false);
    }
    __syncthreads();   // reads of buf[cur] done before it is restaged
  }

  const int m0 = mt * 16;
  const int h = lane >> 4;
#pragma unroll
  for (int j = 0; j < NT; ++j) {
    const int n0 = (nt0 + j) * 16;
    if (n0 >= N) break;                                   // wave-uniform
    const int n = n0 + (lane & 15);
    const float bv = bias ? bias[n] : 0.0f;
#pragma unroll
    for (int r = 0; r < 8; ++r) {
      const int m = m0 + r + h * 8;
      C[(size_t)m * ldc + n] = acc[j][r] + bv;
    }
  }
}

// ------------------------------ mean over P --------------------------------
__global__ void mean_kernel(const float* __restrict__ enc, float* __restrict__ mean) {
  const int b = blockIdx.x;
  const float* e = enc + (size_t)b * PPOS * DENC;
  for (int d = threadIdx.x; d < DENC; d += blockDim.x) {
    float s = 0.0f;
    for (int p = 0; p < PPOS; ++p) s += e[(size_t)p * DENC + d];
    mean[(size_t)b * DENC + d] = s * (1.0f / (float)PPOS);
  }
}

// --------------------- attention: scores -> softmax -> ctx -----------------
__global__ __launch_bounds__(256)
void attention_kernel(const float* __restrict__ enc,
                      const float* __restrict__ enc_att,
                      const float* __restrict__ hid,
                      const float* __restrict__ w_full,
                      const float* __restrict__ b_full,
                      const float* __restrict__ gate,
                      float* __restrict__ ctx,
                      float* __restrict__ aw_out,
                      const long long* __restrict__ lens, int t) {
  __shared__ float s_score[PPOS];
  __shared__ float s_red[256];
  const int b = blockIdx.x, tid = threadIdx.x;
  const int lane = tid & 31, wv = tid >> 5;

  const float* ea = enc_att + (size_t)b * PPOS * ADIM;
  const float* hb = hid + (size_t)b * ADIM;
  for (int p = wv; p < PPOS; p += 8) {
    const float* row = ea + (size_t)p * ADIM;
    float s = 0.0f;
    for (int a = lane; a < ADIM; a += 32)
      s = fmaf(fmaxf(row[a] + hb[a], 0.0f), w_full[a], s);
#pragma unroll
    for (int off = 16; off > 0; off >>= 1) s += __shfl_down(s, off, 32);
    if (lane == 0) s_score[p] = s + b_full[0];
  }
  __syncthreads();

  const float v = (tid < PPOS) ? s_score[tid] : -3.0e38f;
  s_red[tid] = v;
  __syncthreads();
  for (int s = 128; s > 0; s >>= 1) {
    if (tid < s) s_red[tid] = fmaxf(s_red[tid], s_red[tid + s]);
    __syncthreads();
  }
  const float mx = s_red[0];
  __syncthreads();
  const float ex = (tid < PPOS) ? __expf(v - mx) : 0.0f;
  s_red[tid] = ex;
  __syncthreads();
  for (int s = 128; s > 0; s >>= 1) {
    if (tid < s) s_red[tid] += s_red[tid + s];
    __syncthreads();
  }
  const float inv = 1.0f / s_red[0];
  __syncthreads();
  if (tid < PPOS) s_score[tid] = ex * inv;
  __syncthreads();

  const float mask = (t < (int)(lens[b] - 1)) ? 1.0f : 0.0f;
  if (tid < PPOS)
    aw_out[((size_t)b * TSTEP + t) * PPOS + tid] = s_score[tid] * mask;

  const float* eb = enc + (size_t)b * PPOS * DENC;
  const float* gb = gate + (size_t)b * DENC;
  float* cb = ctx + (size_t)b * DENC;
  for (int d = tid; d < DENC; d += 256) {
    float s = 0.0f;
    for (int p = 0; p < PPOS; ++p)
      s = fmaf(s_score[p], eb[(size_t)p * DENC + d], s);
    cb[d] = s * gb[d];
  }
}

// ------------------------- build x = [emb_t | ctx] -------------------------
__global__ void build_x_kernel(const float* __restrict__ emb,
                               const long long* __restrict__ caps, int t,
                               const float* __restrict__ ctx,
                               float* __restrict__ x) {
  const int b = blockIdx.x;
  const long long w = caps[(size_t)b * LCAP + t];
  const float* er = emb + (size_t)w * EDIM;
  float* xb = x + (size_t)b * XDIM;
  for (int i = threadIdx.x; i < EDIM; i += blockDim.x) xb[i] = er[i];
  const float* cb = ctx + (size_t)b * DENC;
  for (int i = threadIdx.x; i < DENC; i += blockDim.x) xb[EDIM + i] = cb[i];
}

// ------------------------------- LSTM cell ---------------------------------
__global__ __launch_bounds__(512)
void lstm_kernel(const float* __restrict__ gates,
                 float* __restrict__ h, float* __restrict__ c,
                 float* __restrict__ h_new,
                 const long long* __restrict__ lens, int t) {
  const int b = blockIdx.x, j = threadIdx.x;   // blockDim = 512
  const float* g = gates + (size_t)b * GDIM;
  const float ig = g[j], fg = g[HDIM + j], gg = g[2 * HDIM + j], og = g[3 * HDIM + j];
  const float co = c[(size_t)b * HDIM + j];
  const float cn = sigmoidf_(fg) * co + sigmoidf_(ig) * tanhf(gg);
  const float hn = sigmoidf_(og) * tanhf(cn);
  h_new[(size_t)b * HDIM + j] = hn;
  if (t < (int)(lens[b] - 1)) {
    h[(size_t)b * HDIM + j] = hn;
    c[(size_t)b * HDIM + j] = cn;
  }
}

// -------------------------- misc small kernels -----------------------------
__global__ void bias_sum_kernel(const float* a, const float* b, float* o, int n) {
  const int i = blockIdx.x * blockDim.x + threadIdx.x;
  if (i < n) o[i] = a[i] + b[i];
}

__global__ void copy_meta_kernel(const long long* __restrict__ caps,
                                 const long long* __restrict__ lens,
                                 float* __restrict__ out_cap,
                                 float* __restrict__ out_len) {
  const int i = blockIdx.x * blockDim.x + threadIdx.x;
  if (i < BSZ * LCAP) out_cap[i] = (float)caps[i];     // exact: ids < 2^24
  if (i < BSZ) out_len[i] = (float)(lens[i] - 1);
}

// ------------------------------ host side ----------------------------------
static void pack_w(const float* W, __bf16* P, int N, int K, hipStream_t s) {
  const int threads = (N / 16) * (K / 32) * 32;
  pack_w_kernel<<<(threads + 255) / 256, 256, 0, s>>>(W, P, N, K);
}
static void pack_a(const float* A, __bf16* P, int M, int K, hipStream_t s) {
  const int threads = (M / 16) * (K / 32) * 32;
  pack_a_kernel<<<(threads + 255) / 256, 256, 0, s>>>(A, P, M, K);
}
static void gemm(const __bf16* Ap, const __bf16* Wp, const float* bias,
                 float* C, int ldc, int M, int N, int K, int act, int accum,
                 const long long* lens, int t, hipStream_t s) {
  dim3 grid((N + 127) / 128, M / 16), block(32, 4);   // 4 waves x 2 x 16
  gemm_wmma_pk<<<grid, block, 0, s>>>(Ap, Wp, bias, C, ldc,
                                      M, N, K, act, accum, lens, t);
}

extern "C" void kernel_launch(void* const* d_in, const int* in_sizes, int n_in,
                              void* d_out, int out_size, void* d_ws, size_t ws_size,
                              hipStream_t stream) {
  const float*     enc    = (const float*)d_in[0];
  const long long* caps   = (const long long*)d_in[1];   // int64 per reference
  const long long* lens   = (const long long*)d_in[2];   // int64 per reference
  const float* emb    = (const float*)d_in[3];
  const float* W_enc  = (const float*)d_in[4];
  const float* b_enc  = (const float*)d_in[5];
  const float* W_hid  = (const float*)d_in[6];
  const float* b_hid  = (const float*)d_in[7];
  const float* w_full = (const float*)d_in[8];
  const float* b_full = (const float*)d_in[9];
  const float* W_ih   = (const float*)d_in[10];
  const float* b_ih   = (const float*)d_in[11];
  const float* W_hh   = (const float*)d_in[12];
  const float* b_hh   = (const float*)d_in[13];
  const float* W_h0   = (const float*)d_in[14];
  const float* b_h0   = (const float*)d_in[15];
  const float* W_c0   = (const float*)d_in[16];
  const float* b_c0   = (const float*)d_in[17];
  const float* W_beta = (const float*)d_in[18];
  const float* b_beta = (const float*)d_in[19];
  const float* W_fc   = (const float*)d_in[20];
  const float* b_fc   = (const float*)d_in[21];

  float* out = (float*)d_out;
  float* out_pred = out;                                  // (B,T,V)
  float* out_cap  = out_pred + (size_t)BSZ * TSTEP * VOC; // (B,L) as float
  float* out_len  = out_cap + (size_t)BSZ * LCAP;         // (B)   as float
  float* out_aw   = out_len + BSZ;                        // (B,T,P)

  // ---- workspace: fp32 region ----
  float* ws      = (float*)d_ws;
  float* w_mean  = ws;                                    // 64*2048
  float* w_h     = w_mean  + (size_t)BSZ * DENC;
  float* w_c     = w_h     + (size_t)BSZ * HDIM;
  float* w_hnew  = w_c     + (size_t)BSZ * HDIM;
  float* w_hid   = w_hnew  + (size_t)BSZ * HDIM;
  float* w_gate  = w_hid   + (size_t)BSZ * ADIM;
  float* w_ctx   = w_gate  + (size_t)BSZ * DENC;
  float* w_x     = w_ctx   + (size_t)BSZ * DENC;
  float* w_gates = w_x     + (size_t)BSZ * XDIM;
  float* w_bsum  = w_gates + (size_t)BSZ * GDIM;
  float* w_eatt  = w_bsum  + GDIM;                        // 64*196*512
  float* f_end   = w_eatt  + (size_t)BSZ * PPOS * ADIM;

  // ---- workspace: packed bf16 region (32B aligned; sizes mult. of 512) ----
  __bf16* bp      = (__bf16*)f_end;
  __bf16* pk_enc  = bp;                                   // 12544*2048
  __bf16* pk_Wenc = pk_enc  + (size_t)BSZ * PPOS * DENC;  // 512*2048
  __bf16* pk_Whid = pk_Wenc + (size_t)ADIM * DENC;        // 512*512
  __bf16* pk_Wbet = pk_Whid + (size_t)ADIM * HDIM;        // 2048*512
  __bf16* pk_Wih  = pk_Wbet + (size_t)DENC * HDIM;        // 2048*2560
  __bf16* pk_Whh  = pk_Wih  + (size_t)GDIM * XDIM;        // 2048*512
  __bf16* pk_Wfc  = pk_Whh  + (size_t)GDIM * HDIM;        // 10000*512
  __bf16* pk_Wh0  = pk_Wfc  + (size_t)VOC * HDIM;         // 512*2048
  __bf16* pk_Wc0  = pk_Wh0  + (size_t)HDIM * DENC;        // 512*2048
  __bf16* pk_mean = pk_Wc0  + (size_t)HDIM * DENC;        // 64*2048
  __bf16* pk_h    = pk_mean + (size_t)BSZ * DENC;         // 64*512
  __bf16* pk_x    = pk_h    + (size_t)BSZ * HDIM;         // 64*2560
  __bf16* pk_hn   = pk_x    + (size_t)BSZ * XDIM;         // 64*512

  // ---- one-time: repack operands to bf16 fragment layout ----
  pack_a(enc, pk_enc, BSZ * PPOS, DENC, stream);
  pack_w(W_enc,  pk_Wenc, ADIM, DENC, stream);
  pack_w(W_hid,  pk_Whid, ADIM, HDIM, stream);
  pack_w(W_beta, pk_Wbet, DENC, HDIM, stream);
  pack_w(W_ih,   pk_Wih,  GDIM, XDIM, stream);
  pack_w(W_hh,   pk_Whh,  GDIM, HDIM, stream);
  pack_w(W_fc,   pk_Wfc,  VOC,  HDIM, stream);
  pack_w(W_h0,   pk_Wh0,  HDIM, DENC, stream);
  pack_w(W_c0,   pk_Wc0,  HDIM, DENC, stream);

  mean_kernel<<<BSZ, 256, 0, stream>>>(enc, w_mean);
  pack_a(w_mean, pk_mean, BSZ, DENC, stream);
  bias_sum_kernel<<<(GDIM + 255) / 256, 256, 0, stream>>>(b_ih, b_hh, w_bsum, GDIM);
  copy_meta_kernel<<<(BSZ * LCAP + 255) / 256, 256, 0, stream>>>(caps, lens, out_cap, out_len);

  // h0 / c0
  gemm(pk_mean, pk_Wh0, b_h0, w_h, HDIM, BSZ, HDIM, DENC, 0, 0, nullptr, 0, stream);
  gemm(pk_mean, pk_Wc0, b_c0, w_c, HDIM, BSZ, HDIM, DENC, 0, 0, nullptr, 0, stream);

  // enc_att = enc(12544x2048) @ W_enc^T + b_enc  (async-LDS staged A)
  {
    dim3 grid((ADIM + 127) / 128, (BSZ * PPOS) / 16), block(32, 4);
    gemm_wmma_pkA_async<<<grid, block, 0, stream>>>(pk_enc, pk_Wenc, b_enc,
                                                    w_eatt, ADIM,
                                                    BSZ * PPOS, ADIM, DENC);
  }

  // ---- recurrence ----
  for (int t = 0; t < TSTEP; ++t) {
    pack_a(w_h, pk_h, BSZ, HDIM, stream);
    gemm(pk_h, pk_Whid, b_hid, w_hid, ADIM, BSZ, ADIM, HDIM, 0, 0, nullptr, 0, stream);
    gemm(pk_h, pk_Wbet, b_beta, w_gate, DENC, BSZ, DENC, HDIM, 1, 0, nullptr, 0, stream);
    attention_kernel<<<BSZ, 256, 0, stream>>>(enc, w_eatt, w_hid, w_full, b_full,
                                              w_gate, w_ctx, out_aw, lens, t);
    build_x_kernel<<<BSZ, 256, 0, stream>>>(emb, caps, t, w_ctx, w_x);
    pack_a(w_x, pk_x, BSZ, XDIM, stream);
    gemm(pk_x, pk_Wih, w_bsum, w_gates, GDIM, BSZ, GDIM, XDIM, 0, 0, nullptr, 0, stream);
    gemm(pk_h, pk_Whh, nullptr, w_gates, GDIM, BSZ, GDIM, HDIM, 0, 1, nullptr, 0, stream);
    lstm_kernel<<<BSZ, 512, 0, stream>>>(w_gates, w_h, w_c, w_hnew, lens, t);
    pack_a(w_hnew, pk_hn, BSZ, HDIM, stream);
    gemm(pk_hn, pk_Wfc, b_fc, out_pred + (size_t)t * VOC, TSTEP * VOC,
         BSZ, VOC, HDIM, 0, 0, lens, t, stream);
  }
}
